// MixtureOfDictionaryExperts_81647328297253
// MI455X (gfx1250) — matrix operand order repaired and checked by reading, not compiled, not run on hardware
//
#include <hip/hip_runtime.h>

#define BATCH    8192
#define IN_DIM   512
#define Q_DIM    128
#define CODE_DIM 1024
#define KEXP     8
#define PROJ_DIM 64
#define NUM_LAYERS 5

typedef __attribute__((ext_vector_type(16))) __bf16 v16bf;
typedef __attribute__((ext_vector_type(8)))  float  v8f;

__constant__ int c_sl[KEXP] = {5, 150, 296, 441, 587, 732, 878, 1024}; // int(np.linspace(5,1024,8))

__device__ __forceinline__ unsigned bf_bits(__bf16 h) {
  union { __bf16 h; unsigned short u; } c; c.h = h; return (unsigned)c.u;
}

// A-operand fragment (16x32 bf16, wave32 layout), lane holds row M = lane&15.
// rowp must already include the per-lane +8*(lane>=16) element offset.
// elements e=0..7  -> K = kbase + 8*(lane>=16) + e
// elements e=8..15 -> K = kbase + 16 + 8*(lane>=16) + (e-8)
__device__ __forceinline__ v16bf load_afrag(const __bf16* rowp, int kbase) {
  union { v16bf v; uint4 q[2]; } u;
  u.q[0] = *(const uint4*)(rowp + kbase);
  u.q[1] = *(const uint4*)(rowp + kbase + 16);
  return u.v;
}

// ---------------------------------------------------------------------------
// Pre-swizzle a [Kdim x Ndim] f32 weight matrix into bf16 B-operand fragments:
// out[(((kb*(N/16)+ct)*32 + lane)*16 + e)] = W[kb*32 + (e/8)*16 + (lane/16)*8 + e%8][ct*16 + lane%16]
// so each wave's B fragment is one contiguous 32-byte load (saddr + voffset + imm).
// ---------------------------------------------------------------------------
__global__ void swizzle_kernel(const float* __restrict__ W, __bf16* __restrict__ outw,
                               int Kdim, int Ndim) {
  long long total = (long long)Kdim * Ndim;
  int nt = Ndim >> 4;
  for (long long i = (long long)blockIdx.x * blockDim.x + threadIdx.x; i < total;
       i += (long long)gridDim.x * blockDim.x) {
    int e = (int)(i & 15);
    long long t = i >> 4;
    int ln = (int)(t & 31);
    t >>= 5;
    int ct = (int)(t % nt);
    int kb = (int)(t / nt);
    int kr = kb * 32 + (e >> 3) * 16 + (ln >> 4) * 8 + (e & 7);
    int nc = ct * 16 + (ln & 15);
    outw[i] = (__bf16)W[(long long)kr * Ndim + nc];
  }
}

// Fold Wq and keys into M = (Wq @ keys^T)/sqrt(Q), c = (bq.keys)/sqrt(Q).
__global__ void gate_prep_kernel(const float* __restrict__ Wq, const float* __restrict__ bq,
                                 const float* __restrict__ keys, float* __restrict__ Mmat,
                                 float* __restrict__ cvec) {
  const float inv = 0.0883883476483184406f; // 1/sqrt(128)
  int i = blockIdx.x * blockDim.x + threadIdx.x;
  if (i < IN_DIM * KEXP) {
    int row = i >> 3, j = i & 7;
    float s = 0.f;
    for (int t = 0; t < Q_DIM; ++t) s += Wq[row * Q_DIM + t] * keys[j * Q_DIM + t];
    Mmat[row * KEXP + j] = s * inv;
  }
  if (i < KEXP) {
    float s = 0.f;
    for (int t = 0; t < Q_DIM; ++t) s += bq[t] * keys[i * Q_DIM + t];
    cvec[i] = s * inv;
  }
}

// Per-row expert selection: smallest k with s_k >= s_max + ln(0.9); build row lists.
// (p_j >= 0.9*p_max  <=>  s_j >= s_max + ln 0.9, so no softmax needed.)
__global__ void gate_kernel(const float* __restrict__ x, const float* __restrict__ Mmat,
                            const float* __restrict__ cvec, int* __restrict__ cnt,
                            int* __restrict__ rowlist) {
  int row = blockIdx.x * blockDim.x + threadIdx.x;
  if (row >= BATCH) return;
  float s[KEXP];
#pragma unroll
  for (int j = 0; j < KEXP; ++j) s[j] = cvec[j];
  const float* xr = x + (size_t)row * IN_DIM;
  for (int i = 0; i < IN_DIM; ++i) {
    float xv = xr[i];
    const float* m = Mmat + i * KEXP;
#pragma unroll
    for (int j = 0; j < KEXP; ++j) s[j] += xv * m[j];
  }
  float smax = s[0];
#pragma unroll
  for (int j = 1; j < KEXP; ++j) smax = fmaxf(smax, s[j]);
  const float thr = smax - 0.105360515657826301f; // + ln(0.9)
  int idx = KEXP - 1;
#pragma unroll
  for (int j = KEXP - 1; j >= 0; --j)
    if (s[j] >= thr) idx = j;
  int slot = atomicAdd(&cnt[idx], 1);
  rowlist[idx * BATCH + slot] = row;
}

// ---------------------------------------------------------------------------
// Fused expert kernel: one workgroup (16 wave32) per 16-row tile of an expert.
// Each wave owns a 16x64 slice (4 column tiles) of the 16x1024 activation.
// ---------------------------------------------------------------------------
__global__ __launch_bounds__(512)
void moe_main_kernel(const float* __restrict__ x,
                     const __bf16* __restrict__ We_sw,
                     const __bf16* __restrict__ S_sw,
                     const __bf16* __restrict__ W1_sw,
                     const __bf16* __restrict__ W2_sw,
                     const float* __restrict__ b1,
                     const float* __restrict__ b2,
                     const float* __restrict__ theta,
                     const int* __restrict__ cnt,
                     const int* __restrict__ rowlist,
                     float* __restrict__ out) {
  const int kexp = blockIdx.y;
  const int tile = blockIdx.x;
  const int count = cnt[kexp];
  if (tile * 16 >= count) return;
  const int rem = (count - tile * 16 < 16) ? (count - tile * 16) : 16;

  __shared__ __align__(16) __bf16 zsh[16][CODE_DIM + 8];        // z (later h), bf16 row-major
  __shared__ __align__(16) unsigned char scratch[16 * 520 * 2]; // xg / histograms / out accum
  __shared__ int rows_sh[16];
  __shared__ unsigned int scan_sh[16];

  const int tid  = threadIdx.x;
  const int wave = tid >> 5;
  const int lane = tid & 31;
  const int mrow = lane & 15;         // A row / C column position
  const int hi8  = (lane >> 4) * 8;
  const float th = theta[kexp];

  // Per-lane B-fragment element offsets; (kb,ct) displacements become IOFFSET imms.
  const int boff64 = (wave * 128 + lane) * 16; // nt=64 matrices (We, S, W1)
  const int boffW2 = lane * 16;                // nt=4 matrix (W2)

  // gather row indices (clamped so WMMA tiles stay full; dup rows never stored)
  if (tid < 16) {
    int idx = tile * 16 + tid;
    if (idx >= count) idx = count - 1;
    rows_sh[tid] = rowlist[kexp * BATCH + idx];
  }
  __syncthreads();

  // gather x rows into LDS as bf16 [16][520]
  __bf16* xg = (__bf16*)scratch;
#pragma unroll
  for (int r = 0; r < 16; ++r)
    xg[r * 520 + tid] = (__bf16)x[(size_t)rows_sh[r] * IN_DIM + tid];
  __syncthreads();

  // ---- Bx = x @ We[k]; each wave keeps its 4 column tiles in registers ----
  v8f bx[4];
#pragma unroll
  for (int ct = 0; ct < 4; ++ct)
#pragma unroll
    for (int r = 0; r < 8; ++r) bx[ct][r] = 0.f;

  {
    const __bf16* WeK = We_sw + (size_t)kexp * IN_DIM * CODE_DIM;
    const __bf16* xrow = xg + mrow * 520 + hi8;
#pragma unroll
    for (int kb = 0; kb < IN_DIM / 32; ++kb) {
      v16bf a = load_afrag(xrow, kb * 32);
#pragma unroll
      for (int ct = 0; ct < 4; ++ct) {
        v16bf b = *(const v16bf*)(WeK + boff64 + kb * 32768 + ct * 512);
        bx[ct] = __builtin_amdgcn_wmma_f32_16x16x32_bf16(false, a, false, b, (short)0,
                                                         bx[ct], false, false);
      }
    }
  }
  __syncthreads(); // xg reads done; scratch reused below

  auto soft_store = [&](const v8f& acc, int ctg) {
    int col = ctg * 16 + mrow;
#pragma unroll
    for (int r = 0; r < 8; ++r) {       // C/D layout: row = r + 8*(lane/16)
      float v = acc[r];
      float a = fabsf(v) - th;
      float s = (a > 0.f) ? ((v > 0.f) ? a : -a) : 0.f;
      zsh[hi8 + r][col] = (__bf16)s;
    }
  };

  // z0 = soft(Bx, theta)
#pragma unroll
  for (int ct = 0; ct < 4; ++ct) soft_store(bx[ct], wave * 4 + ct);
  __syncthreads();

  // ---- 5 x  z = soft(Bx + z @ S[k]) ----
  {
    const __bf16* SK = S_sw + (size_t)kexp * CODE_DIM * CODE_DIM;
    const __bf16* zrow = &zsh[mrow][0] + hi8;
#pragma unroll 1
    for (int it = 0; it < NUM_LAYERS; ++it) {
      v8f acc[4];
#pragma unroll
      for (int ct = 0; ct < 4; ++ct) acc[ct] = bx[ct];
#pragma unroll
      for (int kb = 0; kb < CODE_DIM / 32; ++kb) {
        v16bf a = load_afrag(zrow, kb * 32);
#pragma unroll
        for (int ct = 0; ct < 4; ++ct) {
          v16bf b = *(const v16bf*)(SK + boff64 + kb * 32768 + ct * 512);
          acc[ct] = __builtin_amdgcn_wmma_f32_16x16x32_bf16(false, a, false, b, (short)0,
                                                            acc[ct], false, false);
        }
      }
      __syncthreads(); // all reads of z complete before overwrite
#pragma unroll
      for (int ct = 0; ct < 4; ++ct) soft_store(acc[ct], wave * 4 + ct);
      __syncthreads();
    }
  }

  // ---- exact top-k magnitude threshold: 2-pass radix select on bf16 bits ----
  const int kk = c_sl[kexp];
  if (kk < CODE_DIM) {                       // kexp==7 keeps all 1024 -> skip
    int* hist = (int*)scratch + wave * 256;  // wave-private bins; row = wave
    const __bf16* zr = &zsh[wave][0];
    for (int i = lane; i < 256; i += 32) hist[i] = 0;
    __syncthreads();
    for (int i = lane; i < CODE_DIM; i += 32)
      atomicAdd(&hist[(bf_bits(zr[i]) & 0x7fffu) >> 7], 1);
    __syncthreads();
    if (lane == 0) {
      int cum = 0, b0 = 0, remk = kk;
      for (int b = 255; b >= 0; --b) {
        cum += hist[b];
        if (cum >= kk) { b0 = b; remk = kk - (cum - hist[b]); break; }
      }
      scan_sh[wave] = ((unsigned)b0 << 16) | (unsigned)remk;
    }
    __syncthreads();
    unsigned b0 = scan_sh[wave] >> 16;
    int remk = (int)(scan_sh[wave] & 0xffffu);
    for (int i = lane; i < 128; i += 32) hist[i] = 0;
    __syncthreads();
    for (int i = lane; i < CODE_DIM; i += 32) {
      unsigned p = bf_bits(zr[i]) & 0x7fffu;
      if ((p >> 7) == b0) atomicAdd(&hist[p & 127], 1);
    }
    __syncthreads();
    if (lane == 0) {
      int cum = 0, lo = 0;
      for (int b = 127; b >= 0; --b) {
        cum += hist[b];
        if (cum >= remk) { lo = b; break; }
      }
      scan_sh[wave] = (b0 << 7) | (unsigned)lo; // kth-largest |z| bit pattern
    }
    __syncthreads();
    unsigned kth = scan_sh[wave];
    for (int i = lane; i < CODE_DIM; i += 32)
      if ((bf_bits(zr[i]) & 0x7fffu) < kth) zsh[wave][i] = (__bf16)0.f; // keep az >= kth
  }
  __syncthreads();

  // zero output accumulator (reuses scratch); made visible by syncs below
  float* outsh = (float*)scratch; // [16][68]
  for (int i = tid; i < 16 * 68; i += 512) outsh[i] = 0.f;

  // ---- h = relu(z @ W1 + b1) ----
  v8f hacc[4];
#pragma unroll
  for (int ct = 0; ct < 4; ++ct) {
    float bb = b1[(wave * 4 + ct) * 16 + mrow];
#pragma unroll
    for (int r = 0; r < 8; ++r) hacc[ct][r] = bb;
  }
  {
    const __bf16* zrow = &zsh[mrow][0] + hi8;
#pragma unroll
    for (int kb = 0; kb < CODE_DIM / 32; ++kb) {
      v16bf a = load_afrag(zrow, kb * 32);
#pragma unroll
      for (int ct = 0; ct < 4; ++ct) {
        v16bf b = *(const v16bf*)(W1_sw + boff64 + kb * 32768 + ct * 512);
        hacc[ct] = __builtin_amdgcn_wmma_f32_16x16x32_bf16(false, a, false, b, (short)0,
                                                           hacc[ct], false, false);
      }
    }
  }
  __syncthreads();
#pragma unroll
  for (int ct = 0; ct < 4; ++ct) {
    int col = (wave * 4 + ct) * 16 + mrow;
#pragma unroll
    for (int r = 0; r < 8; ++r)
      zsh[hi8 + r][col] = (__bf16)fmaxf(hacc[ct][r], 0.f);
  }
  __syncthreads();

  // ---- out = h @ W2 + b2 : split K over the 16 waves, reduce in LDS ----
  v8f oacc[4];
#pragma unroll
  for (int ct = 0; ct < 4; ++ct)
#pragma unroll
    for (int r = 0; r < 8; ++r) oacc[ct][r] = 0.f;
  {
    const __bf16* zrow = &zsh[mrow][0] + hi8;
#pragma unroll
    for (int s = 0; s < 2; ++s) {
      int kb = wave * 2 + s;
      v16bf a = load_afrag(zrow, kb * 32);
#pragma unroll
      for (int ct = 0; ct < 4; ++ct) {
        v16bf b = *(const v16bf*)(W2_sw + boffW2 + kb * 2048 + ct * 512);
        oacc[ct] = __builtin_amdgcn_wmma_f32_16x16x32_bf16(false, a, false, b, (short)0,
                                                           oacc[ct], false, false);
      }
    }
  }
#pragma unroll
  for (int ct = 0; ct < 4; ++ct)
#pragma unroll
    for (int r = 0; r < 8; ++r)
      atomicAdd(&outsh[(hi8 + r) * 68 + ct * 16 + mrow], oacc[ct][r]);
  __syncthreads();

  for (int i = tid; i < 16 * PROJ_DIM; i += 512) {
    int r = i >> 6, c = i & 63;
    if (r < rem)
      out[(size_t)rows_sh[r] * PROJ_DIM + c] = outsh[r * 68 + c] + b2[c];
  }
}

// ---------------------------------------------------------------------------
extern "C" void kernel_launch(void* const* d_in, const int* in_sizes, int n_in,
                              void* d_out, int out_size, void* d_ws, size_t ws_size,
                              hipStream_t stream) {
  (void)in_sizes; (void)n_in; (void)out_size; (void)ws_size;
  const float* x     = (const float*)d_in[0];
  const float* Wq    = (const float*)d_in[1];
  const float* bq    = (const float*)d_in[2];
  const float* keys  = (const float*)d_in[3];
  const float* We    = (const float*)d_in[4];
  const float* S     = (const float*)d_in[5];
  const float* theta = (const float*)d_in[6];
  const float* W1    = (const float*)d_in[7];
  const float* b1    = (const float*)d_in[8];
  const float* W2    = (const float*)d_in[9];
  const float* b2    = (const float*)d_in[10];
  float* out = (float*)d_out;

  char* ws = (char*)d_ws;
  size_t off = 0;
  auto carve = [&](size_t bytes) { size_t o = off; off = (off + bytes + 255) & ~(size_t)255; return o; };
  __bf16* We_sw = (__bf16*)(ws + carve((size_t)KEXP * IN_DIM * CODE_DIM * 2));
  __bf16* S_sw  = (__bf16*)(ws + carve((size_t)KEXP * CODE_DIM * CODE_DIM * 2));
  __bf16* W1_sw = (__bf16*)(ws + carve((size_t)CODE_DIM * CODE_DIM * 2));
  __bf16* W2_sw = (__bf16*)(ws + carve((size_t)CODE_DIM * PROJ_DIM * 2));
  float*  Mmat  = (float*)(ws + carve((size_t)IN_DIM * KEXP * 4));
  float*  cvec  = (float*)(ws + carve((size_t)KEXP * 4));
  int*    cnt   = (int*)(ws + carve((size_t)KEXP * 4));
  int*    rlist = (int*)(ws + carve((size_t)KEXP * BATCH * 4));

  // weight pre-swizzle into WMMA B-operand layout (bf16), every call (deterministic)
  for (int k = 0; k < KEXP; ++k)
    swizzle_kernel<<<256, 256, 0, stream>>>(We + (size_t)k * IN_DIM * CODE_DIM,
                                            We_sw + (size_t)k * IN_DIM * CODE_DIM,
                                            IN_DIM, CODE_DIM);
  for (int k = 0; k < KEXP; ++k)
    swizzle_kernel<<<512, 256, 0, stream>>>(S + (size_t)k * CODE_DIM * CODE_DIM,
                                            S_sw + (size_t)k * CODE_DIM * CODE_DIM,
                                            CODE_DIM, CODE_DIM);
  swizzle_kernel<<<512, 256, 0, stream>>>(W1, W1_sw, CODE_DIM, CODE_DIM);
  swizzle_kernel<<<64, 256, 0, stream>>>(W2, W2_sw, CODE_DIM, PROJ_DIM);

  gate_prep_kernel<<<(IN_DIM * KEXP + 255) / 256, 256, 0, stream>>>(Wq, bq, keys, Mmat, cvec);
  hipMemsetAsync(cnt, 0, KEXP * sizeof(int), stream);
  gate_kernel<<<BATCH / 256, 256, 0, stream>>>(x, Mmat, cvec, cnt, rlist);

  dim3 grid(BATCH / 16, KEXP);
  moe_main_kernel<<<grid, 512, 0, stream>>>(x, We_sw, S_sw, W1_sw, W2_sw,
                                            b1, b2, theta, cnt, rlist, out);
}
